// ButterflyLayer_59322088292569
// MI455X (gfx1250) — matrix operand order repaired and checked by reading, not compiled, not run on hardware
//
#include <hip/hip_runtime.h>
#include <hip/hip_bf16.h>

typedef __attribute__((ext_vector_type(16))) _Float16 v16h;
typedef __attribute__((ext_vector_type(8)))  _Float16 v8h;
typedef __attribute__((ext_vector_type(8)))  float    v8f;

#define BATCH 16384
#define DIM   2048
#define LOGD  11
#define MAXBLK 1024

#define BM 128
#define BN 128
#define BK 32
#define LDT 40   // LDS row stride in halfs (32 + 8 pad; 80B rows keep 16B alignment)

// Low 32 bits of a generic pointer to LDS == wave-relative LDS byte address.
__device__ __forceinline__ unsigned lds_off32(const void* p) {
  return (unsigned)(unsigned long long)(uintptr_t)p;
}

// CDNA5 async copy: global -> LDS, 16B per lane, tracked by ASYNCcnt.
__device__ __forceinline__ void async_g2l_b128(unsigned lds, const void* gaddr) {
  asm volatile("global_load_async_to_lds_b128 %0, %1, off"
               :: "v"(lds), "v"(gaddr) : "memory");
}

// Indivisible pipeline-sync point: drain this wave's async fills and LDS reads,
// then workgroup split-barrier. Single asm blob so the scheduler cannot rotate
// LDS traffic into the middle of the protocol.
__device__ __forceinline__ void pipeline_barrier() {
#if __has_builtin(__builtin_amdgcn_sched_barrier)
  __builtin_amdgcn_sched_barrier(0);
#endif
  asm volatile(
      "s_wait_asynccnt 0\n\t"
      "s_wait_dscnt 0\n\t"
      "s_barrier_signal -1\n\t"
      "s_barrier_wait -1"
      ::: "memory");
#if __has_builtin(__builtin_amdgcn_sched_barrier)
  __builtin_amdgcn_sched_barrier(0);
#endif
}

// ---------------- f32 -> f16 conversion (b128 in / b128 out) -----------------
__global__ __launch_bounds__(256) void cvt_f32_f16(const float* __restrict__ in,
                                                   _Float16* __restrict__ out,
                                                   int n8) {
  int i = blockIdx.x * 256 + threadIdx.x;
  if (i >= n8) return;
  const float4* p = (const float4*)in + (size_t)i * 2;
  float4 lo = p[0], hi = p[1];
  v8h o;
  o[0] = (_Float16)lo.x; o[1] = (_Float16)lo.y; o[2] = (_Float16)lo.z; o[3] = (_Float16)lo.w;
  o[4] = (_Float16)hi.x; o[5] = (_Float16)hi.y; o[6] = (_Float16)hi.z; o[7] = (_Float16)hi.w;
  *((v8h*)out + i) = o;
}

// ---------------- WMMA GEMM: C[M,N] = A[M,K] @ B[N,K]^T (+bias) --------------
// A, B row-major f16, K contiguous. F32OUT=true: f32 + bias -> Cf; else f16 -> Ch.
template <bool F32OUT>
__global__ __launch_bounds__(256, 1) void gemm_wmma(const _Float16* __restrict__ A,
                                                    const _Float16* __restrict__ B,
                                                    float* __restrict__ Cf,
                                                    _Float16* __restrict__ Ch,
                                                    const float* __restrict__ bias,
                                                    int M, int N, int K) {
  __shared__ _Float16 sA[2][BM * LDT];
  __shared__ _Float16 sB[2][BN * LDT];

  const int t    = threadIdx.x;
  const int lane = t & 31;
  const int wave = t >> 5;
  const int wm   = wave >> 1;   // 0..3 -> 32-row strip
  const int wn   = wave & 1;    // 0..1 -> 64-col strip
  const int l16  = lane & 15;
  const int lhi  = lane >> 4;
  const int mBase = blockIdx.y * BM;
  const int nBase = blockIdx.x * BN;

  // Each thread moves 2 x 16B chunks per matrix per k-tile.
  int rowc[2], kcc[2];
#pragma unroll
  for (int i = 0; i < 2; ++i) {
    int c = t * 2 + i;          // 512 chunks; 4 chunks (64B) per 32-half row
    rowc[i] = c >> 2;
    kcc[i]  = c & 3;
  }

  // Precompute per-thread global bases (bytes) and LDS destinations.
  const char* aPtr[2];
  const char* bPtr[2];
  unsigned ldsA[2][2], ldsB[2][2];
#pragma unroll
  for (int i = 0; i < 2; ++i) {
    aPtr[i] = (const char*)(A + (size_t)(mBase + rowc[i]) * K + kcc[i] * 8);
    bPtr[i] = (const char*)(B + (size_t)(nBase + rowc[i]) * K + kcc[i] * 8);
#pragma unroll
    for (int buf = 0; buf < 2; ++buf) {
      ldsA[buf][i] = lds_off32(&sA[buf][rowc[i] * LDT + kcc[i] * 8]);
      ldsB[buf][i] = lds_off32(&sB[buf][rowc[i] * LDT + kcc[i] * 8]);
    }
  }

  v8f acc[2][4];
#pragma unroll
  for (int f = 0; f < 2; ++f)
#pragma unroll
    for (int j = 0; j < 4; ++j) acc[f][j] = (v8f)0.0f;

  auto computeTile = [&](int cur) {
    v16h af[2], bf[4];
#pragma unroll
    for (int f = 0; f < 2; ++f) {
      const int r  = wm * 32 + f * 16 + l16;
      const int k0 = lhi * 8;
      v8h lo = *(const v8h*)&sA[cur][r * LDT + k0];
      v8h hi = *(const v8h*)&sA[cur][r * LDT + k0 + 16];
#pragma unroll
      for (int q = 0; q < 8; ++q) { af[f][q] = lo[q]; af[f][q + 8] = hi[q]; }
    }
#pragma unroll
    for (int j = 0; j < 4; ++j) {
      const int n  = wn * 64 + j * 16 + l16;
      const int k0 = lhi * 8;
      v8h lo = *(const v8h*)&sB[cur][n * LDT + k0];
      v8h hi = *(const v8h*)&sB[cur][n * LDT + k0 + 16];
#pragma unroll
      for (int q = 0; q < 8; ++q) { bf[j][q] = lo[q]; bf[j][q + 8] = hi[q]; }
    }
#pragma unroll
    for (int f = 0; f < 2; ++f)
#pragma unroll
      for (int j = 0; j < 4; ++j)
        acc[f][j] = __builtin_amdgcn_wmma_f32_16x16x32_f16(
            false, af[f], false, bf[j], (short)0, acc[f][j], false, false);
  };

  const int KT = K / BK;
  const size_t kstep = (size_t)BK * sizeof(_Float16);

  // Prologue: tile 0 -> buf 0 via async copy.
#pragma unroll
  for (int i = 0; i < 2; ++i) {
    async_g2l_b128(ldsA[0][i], aPtr[i]);
    async_g2l_b128(ldsB[0][i], bPtr[i]);
  }
  pipeline_barrier();

  // Steady state: issue async copy of kt+1 into the other buffer, compute kt.
  for (int kt = 0; kt < KT - 1; ++kt) {
    const int cur = kt & 1;
    const int nxt = cur ^ 1;
    const size_t koff = (size_t)(kt + 1) * kstep;
#pragma unroll
    for (int i = 0; i < 2; ++i) {
      async_g2l_b128(ldsA[nxt][i], aPtr[i] + koff);
      async_g2l_b128(ldsB[nxt][i], bPtr[i] + koff);
    }
    computeTile(cur);
    pipeline_barrier();
  }
  computeTile((KT - 1) & 1);

  // Epilogue per 16x16 tile: lane L, vgpr v -> m = (L/16)*8 + v, n = L%16.
#pragma unroll
  for (int f = 0; f < 2; ++f) {
#pragma unroll
    for (int j = 0; j < 4; ++j) {
      const int n = nBase + wn * 64 + j * 16 + l16;
      const float bv = F32OUT ? bias[n] : 0.0f;
#pragma unroll
      for (int v = 0; v < 8; ++v) {
        const int m = mBase + wm * 32 + f * 16 + lhi * 8 + v;
        if (F32OUT) Cf[(size_t)m * N + n] = acc[f][j][v] + bv;
        else        Ch[(size_t)m * N + n] = (_Float16)acc[f][j][v];
      }
    }
  }
}

// ---------------- Butterfly: 11 Givens-rotation levels, one row per block ----
__global__ __launch_bounds__(256) void butterfly_kernel(_Float16* __restrict__ h,
                                                        const float* __restrict__ a_pad,
                                                        const float* __restrict__ b_pad) {
  __shared__ float row[DIM];
  _Float16* hp = h + (size_t)blockIdx.x * DIM;

  for (int e = threadIdx.x; e < DIM; e += 256) row[e] = (float)hp[e];
  __syncthreads();

#pragma unroll
  for (int l = 0; l < LOGD; ++l) {
    const int bs = 1 << l;
    for (int p = threadIdx.x; p < MAXBLK; p += 256) {
      const int blk = p >> l;
      const int j   = p & (bs - 1);
      const int i0  = blk * (bs << 1) + j;
      const int i1  = i0 + bs;
      const float a = a_pad[l * MAXBLK + blk];
      const float b = b_pad[l * MAXBLK + blk];
      const float x0 = row[i0], x1 = row[i1];
      row[i0] =  a * x0 + b * x1;
      row[i1] = -b * x0 + a * x1;
    }
    __syncthreads();
  }

  for (int e = threadIdx.x; e < DIM; e += 256) hp[e] = (_Float16)row[e];
}

// -----------------------------------------------------------------------------
extern "C" void kernel_launch(void* const* d_in, const int* in_sizes, int n_in,
                              void* d_out, int out_size, void* d_ws, size_t ws_size,
                              hipStream_t stream) {
  const float* x     = (const float*)d_in[0];
  const float* w_in  = (const float*)d_in[1];
  const float* w_out = (const float*)d_in[2];
  const float* b_out = (const float*)d_in[3];
  const float* a_pad = (const float*)d_in[4];
  const float* b_pad = (const float*)d_in[5];
  float* out = (float*)d_out;

  char* ws = (char*)d_ws;
  _Float16* x16 = (_Float16*)ws;                                   // 16384*2048
  _Float16* w1  = (_Float16*)(ws + (size_t)BATCH * DIM * 2);       // 2048*2048
  _Float16* w2  = w1 + (size_t)DIM * DIM;                          // 2048*2048
  _Float16* h16 = w2 + (size_t)DIM * DIM;                          // 16384*2048

  const int nx = BATCH * DIM / 8;
  const int nw = DIM * DIM / 8;
  cvt_f32_f16<<<(nx + 255) / 256, 256, 0, stream>>>(x, x16, nx);
  cvt_f32_f16<<<(nw + 255) / 256, 256, 0, stream>>>(w_in, w1, nw);
  cvt_f32_f16<<<(nw + 255) / 256, 256, 0, stream>>>(w_out, w2, nw);

  dim3 grid(DIM / BN, BATCH / BM);
  gemm_wmma<false><<<grid, 256, 0, stream>>>(x16, w1, nullptr, h16, nullptr,
                                             BATCH, DIM, DIM);
  butterfly_kernel<<<BATCH, 256, 0, stream>>>(h16, a_pad, b_pad);
  gemm_wmma<true><<<grid, 256, 0, stream>>>(h16, w2, out, nullptr, b_out,
                                            BATCH, DIM, DIM);
}
